// FrameTransformerEncoder_8581344658023
// MI455X (gfx1250) — matrix operand order repaired and checked by reading, not compiled, not run on hardware
//
#include <hip/hip_runtime.h>
#include <hip/hip_bf16.h>
#include <stdint.h>

// ---------------------------------------------------------------------------
// FrameTransformer encoder layer for MI455X (gfx1250, wave32, WMMA).
// All large matmuls run through v_wmma_f32_16x16x32_bf16 (bf16 in, f32 acc).
// Global->LDS staging: double-buffered CDNA5 async-to-LDS copies (ASYNCcnt)
// overlapped with the WMMA chain, plus speculative prefetch of tile t+2.
// ---------------------------------------------------------------------------

typedef __attribute__((ext_vector_type(16))) __bf16 v16bf;
typedef __attribute__((ext_vector_type(8)))  float  v8f;

#define BB     8
#define CC     16
#define BINS   1024
#define WW     512
#define FF     2048
#define NBANDS 4
#define MROWS  (BB * WW)          // 4096 token rows
#define EPSV   1e-5f

__device__ __forceinline__ unsigned short f2bf(float f) {
  unsigned u = __float_as_uint(f);
  unsigned r = ((u >> 16) & 1u) + 0x7FFFu;   // round-to-nearest-even
  return (unsigned short)((u + r) >> 16);
}
__device__ __forceinline__ float bf2f(unsigned short h) {
  return __uint_as_float(((unsigned)h) << 16);
}

// low 32 bits of a generic pointer to __shared__ == LDS byte offset (ISA 10.2)
__device__ __forceinline__ unsigned lds_addr32(const void* p) {
  return (unsigned)(unsigned long long)p;
}

// ---------------- block-wide 2-value reduction (256 threads) ----------------
__device__ __forceinline__ void block_reduce2_256(float& s1, float& s2) {
  __shared__ float r1[256];
  __shared__ float r2[256];
  int t = threadIdx.x;
  r1[t] = s1; r2[t] = s2;
  __syncthreads();
  for (int s = 128; s > 0; s >>= 1) {
    if (t < s) { r1[t] += r1[t + s]; r2[t] += r2[t + s]; }
    __syncthreads();
  }
  s1 = r1[0]; s2 = r2[0];
  __syncthreads();
}

// ===========================================================================
// Tiled bf16 WMMA GEMM:  C[M,N] = alpha * op(A[M,K] x B)    (f32 accumulate)
//   NT=false: B is [K,N] row-major.  NT=true: B is [N,K] row-major (C=A*B^T).
//   Batched via blockIdx.z: z = zb*zmod + zn, offsets zb*s?b + zn*s?n (elems).
//   Requires M%128==0, N%64==0, K%32==0 (true for every call site here).
// Block: 256 threads = 8 waves; block tile 128x64; wave tile 32x32 (2x2 WMMA).
// Double-buffered K pipeline: tile t+1 staged (async global->LDS, ASYNCcnt)
// while tile t runs on the matrix cores. One barrier per iteration: buffer b
// is only re-staged in iteration t+1, after the barrier that proves all waves
// finished reading it in iteration t.  NN B tiles need a transpose, so they
// go through VGPRs + ds_store scatter instead of the async path.
// ===========================================================================
#define MBLK 128
#define NBLK 64
#define AS_LD 40   // halves; row byte stride 80 (16B aligned)
#define BS_LD 40

union FragBF { uint4 q[2]; v16bf v; };

template <bool NT, bool RELU, bool OUTBF>
__global__ __launch_bounds__(256)
void gemm_bf16(const unsigned short* __restrict__ A,
               const unsigned short* __restrict__ Bm,
               void* __restrict__ Cp,
               int M, int N, int K, int lda, int ldb, int ldc, float alpha,
               long sAb, long sAn, long sBb, long sBn, long sCb, long sCn,
               int zmod)
{
  __shared__ __attribute__((aligned(16))) unsigned short As[2][MBLK * AS_LD];
  __shared__ __attribute__((aligned(16))) unsigned short Bs[2][NBLK * BS_LD]; // [n][k]

  int z  = blockIdx.z;
  int zb = z / zmod, zn = z % zmod;
  A  += (long)zb * sAb + (long)zn * sAn;
  Bm += (long)zb * sBb + (long)zn * sBn;
  long coff = (long)zb * sCb + (long)zn * sCn;

  const int tid   = threadIdx.x;
  const int lane  = tid & 31;
  const int wv    = tid >> 5;
  const int waveM = wv >> 1;      // 0..3
  const int waveN = wv & 1;       // 0..1
  const int m0 = blockIdx.y * MBLK;
  const int n0 = blockIdx.x * NBLK;

  const int m_in = lane & 15;
  const int hi   = lane >> 4;     // 0/1
  const int kbA  = hi * 8;        // A frag K base (lanes 16-31 -> K=8..)
  const int kbB  = hi * 8;        // B frag K base

  v8f acc[2][2];
  for (int i = 0; i < 2; i++)
    for (int j = 0; j < 2; j++)
      for (int r = 0; r < 8; r++) acc[i][j][r] = 0.0f;

  // per-thread staging geometry (LDS destinations are K-loop invariant)
  const int arow = tid >> 1, acol = (tid & 1) * 16;
  const unsigned short* gpA = A + (long)(m0 + arow) * lda + acol;
  const unsigned ldsA0[2] = { lds_addr32(&As[0][arow * AS_LD + acol]),
                              lds_addr32(&As[1][arow * AS_LD + acol]) };
  const int nrNT = tid >> 2, kcNT = (tid & 3) * 8;
  const unsigned short* gpBnt = Bm + (long)(n0 + nrNT) * ldb + kcNT;
  const unsigned ldsBv[2] = { lds_addr32(&Bs[0][nrNT * BS_LD + kcNT]),
                              lds_addr32(&Bs[1][nrNT * BS_LD + kcNT]) };
  const int krNN = tid >> 3, ncNN = (tid & 7) * 8;
  const unsigned short* gpBnn = Bm + (long)krNN * ldb + n0 + ncNN;

#define STAGE_TILE(tt, buf)                                                   \
  do {                                                                        \
    const unsigned short* pa = gpA + (long)(tt) * 32;                         \
    unsigned long long ga = (unsigned long long)pa;                           \
    asm volatile("global_load_async_to_lds_b128 %0, %1, off"                  \
                 :: "v"(ldsA0[buf]), "v"(ga) : "memory");                     \
    asm volatile("global_load_async_to_lds_b128 %0, %1, off"                  \
                 :: "v"(ldsA0[buf] + 16u), "v"(ga + 16ULL) : "memory");       \
    if (NT) {                                                                 \
      const unsigned short* pb = gpBnt + (long)(tt) * 32;                     \
      asm volatile("global_load_async_to_lds_b128 %0, %1, off"                \
                   :: "v"(ldsBv[buf]), "v"((unsigned long long)pb)            \
                   : "memory");                                               \
    } else {                                                                  \
      const unsigned short* pb = gpBnn + (long)(tt) * 32 * ldb;               \
      uint4 vv = *(const uint4*)pb;                                           \
      const unsigned short* pv = (const unsigned short*)&vv;                  \
      for (int j = 0; j < 8; j++)                                             \
        Bs[buf][(ncNN + j) * BS_LD + krNN] = pv[j];                           \
    }                                                                         \
  } while (0)

  const int T = K >> 5;             // K tiles of 32
  STAGE_TILE(0, 0);                 // pipeline prologue

  for (int t = 0; t < T; t++) {
    const int buf = t & 1;
    asm volatile("s_wait_asynccnt 0x0" ::: "memory");  // tile t resident
    __syncthreads();
    if (t + 1 < T) {
      STAGE_TILE(t + 1, buf ^ 1);                      // overlap with WMMA
      __builtin_prefetch(gpA + (long)(t + 2) * 32, 0, 1);  // speculative
    }

    // ---- per-lane fragments per documented 16-bit A/B striping ----
    FragBF a[2], b[2];
    for (int mt = 0; mt < 2; mt++) {
      int r = waveM * 32 + mt * 16 + m_in;
      a[mt].q[0] = *(const uint4*)&As[buf][r * AS_LD + kbA];      // K kbA..+7
      a[mt].q[1] = *(const uint4*)&As[buf][r * AS_LD + kbA + 16]; // K kbA+16..
    }
    for (int nt = 0; nt < 2; nt++) {
      int n = waveN * 32 + nt * 16 + m_in;
      b[nt].q[0] = *(const uint4*)&Bs[buf][n * BS_LD + kbB];      // K kbB..+7
      b[nt].q[1] = *(const uint4*)&Bs[buf][n * BS_LD + kbB + 16]; // K kbB+16..
    }
    for (int mt = 0; mt < 2; mt++)
      for (int nt = 0; nt < 2; nt++)
        acc[mt][nt] = __builtin_amdgcn_wmma_f32_16x16x32_bf16(
            false, a[mt].v, false, b[nt].v, (short)0, acc[mt][nt],
            false, false);
    // no trailing barrier: iteration t+1's wait+barrier guards buffer reuse
  }
#undef STAGE_TILE

  // ---- epilogue: C/D layout -> lane(n = lane&15), VGPR r -> M = r + 8*hi ----
  for (int mt = 0; mt < 2; mt++)
    for (int nt = 0; nt < 2; nt++) {
      int gn = n0 + waveN * 32 + nt * 16 + m_in;
      for (int r = 0; r < 8; r++) {
        int gm = m0 + waveM * 32 + mt * 16 + r + hi * 8;
        float v = acc[mt][nt][r] * alpha;
        if (RELU) v = v > 0.0f ? v : 0.0f;
        long off = coff + (long)gm * ldc + gn;
        if (OUTBF) ((unsigned short*)Cp)[off] = f2bf(v);
        else       ((float*)Cp)[off] = v;
      }
    }
}

// ===========================================================================
// Elementwise / reduction kernels
// ===========================================================================
__global__ void zero_stats_kernel(float* stats) {
  if (threadIdx.x == 0) { stats[0] = 0.f; stats[1] = 0.f; stats[2] = 0.f; stats[3] = 0.f; }
}

// y[b,i,w] = sum_c x[b,c,i,w]*bl_w[c]; accumulate sum/sumsq for batch-norm
__global__ __launch_bounds__(256)
void bottleneck_kernel(const float* __restrict__ x, const float* __restrict__ bl_w,
                       float* __restrict__ y, float* stats) {
  long idx = (long)blockIdx.x * 256 + threadIdx.x;   // 0..4M-1 over (b,i,w)
  long b   = idx >> 19;                              // BINS*WW = 2^19
  long rem = idx & 524287;
  const float* xp = x + b * (16L * 524288L) + rem;
  float acc = 0.f;
  for (int c = 0; c < CC; c++) acc += xp[(long)c * 524288L] * bl_w[c];
  y[idx] = acc;
  float s = acc, s2 = acc * acc;
  block_reduce2_256(s, s2);
  if (threadIdx.x == 0) { atomicAdd(&stats[0], s); atomicAdd(&stats[1], s2); }
}

__global__ void bn_finalize_kernel(float* stats, const float* bn_w, const float* bn_b) {
  if (threadIdx.x == 0) {
    const float n = 4194304.f;
    float mean = stats[0] / n;
    float var  = stats[1] / n - mean * mean;
    float sc   = rsqrtf(var + EPSV) * bn_w[0];
    stats[2] = sc;
    stats[3] = bn_b[0] - mean * sc;
  }
}

// xt[b,w,i] = relu(y[b,i,w]*scale + shift)  (f32 + bf16 copies)
__global__ __launch_bounds__(256)
void bn_apply_kernel(const float* __restrict__ y, const float* stats,
                     float* __restrict__ xt, unsigned short* __restrict__ xtbf) {
  long idx = (long)blockIdx.x * 256 + threadIdx.x;
  long b = idx >> 19; long rem = idx & 524287;
  int  i = (int)(rem >> 9), w = (int)(rem & 511);
  float v = y[idx] * stats[2] + stats[3];
  v = v > 0.f ? v : 0.f;
  long mo = ((long)(b * WW + w)) * BINS + i;
  xt[mo] = v;
  xtbf[mo] = f2bf(v);
}

// xt = LN(xt + a*sigmoid(b)) with g=[a|b] rows of 2048
__global__ __launch_bounds__(256)
void ln1_kernel(float* __restrict__ xt, unsigned short* __restrict__ xtbf,
                const float* __restrict__ g, const float* w, const float* b) {
  int m = blockIdx.x;
  const float* gr = g + (long)m * 2048;
  float* xr = xt + (long)m * 1024;
  float s = 0.f, s2 = 0.f;
  for (int d = threadIdx.x; d < 1024; d += 256) {
    float av = gr[d], bv = gr[1024 + d];
    float v = xr[d] + av * (1.f / (1.f + __expf(-bv)));
    xr[d] = v; s += v; s2 += v * v;
  }
  block_reduce2_256(s, s2);
  float mean = s * (1.f / 1024.f);
  float var  = s2 * (1.f / 1024.f) - mean * mean;
  float inv  = rsqrtf(var + EPSV);
  for (int d = threadIdx.x; d < 1024; d += 256) {
    float v = (xr[d] - mean) * inv * w[d] + b[d];
    xr[d] = v;
    xtbf[(long)m * 1024 + d] = f2bf(v);
  }
}

// h(bf16) = LN_4096(hL + pad(hR to first 1024 cols))
__global__ __launch_bounds__(256)
void ln2_kernel(const float* __restrict__ hL, const float* __restrict__ hR,
                unsigned short* __restrict__ hbf, const float* w, const float* b) {
  int m = blockIdx.x;
  const float* lr = hL + (long)m * 4096;
  const float* rr = hR + (long)m * 1024;
  float s = 0.f, s2 = 0.f;
  for (int d = threadIdx.x; d < 4096; d += 256) {
    float v = lr[d] + (d < 1024 ? rr[d] : 0.f);
    s += v; s2 += v * v;
  }
  block_reduce2_256(s, s2);
  float mean = s * (1.f / 4096.f);
  float var  = s2 * (1.f / 4096.f) - mean * mean;
  float inv  = rsqrtf(var + EPSV);
  for (int d = threadIdx.x; d < 4096; d += 256) {
    float v = lr[d] + (d < 1024 ? rr[d] : 0.f);
    hbf[(long)m * 4096 + d] = f2bf((v - mean) * inv * w[d] + b[d]);
  }
}

// xt = LN(xt + add)   (ln3 / ln4)
__global__ __launch_bounds__(256)
void ln_add_kernel(float* __restrict__ xt, unsigned short* __restrict__ xtbf,
                   const float* __restrict__ add, const float* w, const float* b) {
  int m = blockIdx.x;
  float* xr = xt + (long)m * 1024;
  const float* ar = add + (long)m * 1024;
  float s = 0.f, s2 = 0.f;
  for (int d = threadIdx.x; d < 1024; d += 256) {
    float v = xr[d] + ar[d];
    xr[d] = v; s += v; s2 += v * v;
  }
  block_reduce2_256(s, s2);
  float mean = s * (1.f / 1024.f);
  float var  = s2 * (1.f / 1024.f) - mean * mean;
  float inv  = rsqrtf(var + EPSV);
  for (int d = threadIdx.x; d < 1024; d += 256) {
    float v = (xr[d] - mean) * inv * w[d] + b[d];
    xr[d] = v;
    xtbf[(long)m * 1024 + d] = f2bf(v);
  }
}

// out[b,0,d,w] = LN(xt + h4)[b,w,d]  (final transpose-scatter)
__global__ __launch_bounds__(256)
void ln5_kernel(const float* __restrict__ xt, const float* __restrict__ add,
                const float* w, const float* b, float* __restrict__ out) {
  int m = blockIdx.x;
  int bb = m >> 9, ww = m & 511;
  const float* xr = xt + (long)m * 1024;
  const float* ar = add + (long)m * 1024;
  float s = 0.f, s2 = 0.f;
  for (int d = threadIdx.x; d < 1024; d += 256) {
    float v = xr[d] + ar[d]; s += v; s2 += v * v;
  }
  block_reduce2_256(s, s2);
  float mean = s * (1.f / 1024.f);
  float var  = s2 * (1.f / 1024.f) - mean * mean;
  float inv  = rsqrtf(var + EPSV);
  for (int d = threadIdx.x; d < 1024; d += 256) {
    float v = (xr[d] + ar[d] - mean) * inv * w[d] + b[d];
    out[((long)bb * 1024 + d) * 512 + ww] = v;
  }
}

// im2col for conv1d k=3 pad=1 over W:  xcol[m, t*1024+i] = xtbf[b, w+t-1, i]
__global__ __launch_bounds__(256)
void im2col_kernel(const unsigned short* __restrict__ xtbf,
                   unsigned short* __restrict__ xcol) {
  long idx = (long)blockIdx.x * 256 + threadIdx.x;   // over 4M (m,i)
  int i = (int)(idx & 1023);
  long m = idx >> 10;
  int b = (int)(m >> 9), w = (int)(m & 511);
  for (int t = 0; t < 3; t++) {
    int ws = w + t - 1;
    unsigned short v = (ws >= 0 && ws < 512)
        ? xtbf[((long)(b * WW + ws) << 10) + i] : (unsigned short)0;
    xcol[m * 3072 + t * 1024 + i] = v;
  }
}

// depthwise conv k=9 pad=4 over W, per channel c of 4096
__global__ __launch_bounds__(256)
void depthwise_kernel(const unsigned short* __restrict__ h,
                      const float* __restrict__ c2d,
                      unsigned short* __restrict__ hc) {
  long idx = (long)blockIdx.x * 256 + threadIdx.x;   // over 16M (m,c)
  int c = (int)(idx & 4095);
  long m = idx >> 12;
  int b = (int)(m >> 9), w = (int)(m & 511);
  float acc = 0.f;
  for (int t = 0; t < 9; t++) {
    int ws = w + t - 4;
    if (ws >= 0 && ws < 512)
      acc += bf2f(h[((long)(b * WW + ws) << 12) + c]) * c2d[c * 9 + t];
  }
  hc[idx] = f2bf(acc);
}

// logits = qk + qe^T + |i-j|/2 * dist_w[n,i]; row softmax -> bf16
__global__ __launch_bounds__(256)
void attn_softmax_kernel(const float* __restrict__ qk, const float* __restrict__ qe,
                         const float* __restrict__ dist_w,
                         unsigned short* __restrict__ attnA) {
  int i = blockIdx.x;           // row  0..511
  int z = blockIdx.y;           // (b,n) 0..31, n = z & 3
  int n = z & 3;
  const float* qkr = qk + (long)z * 262144 + (long)i * 512;
  const float* qec = qe + (long)z * 262144 + i;     // qe[z][j][i], stride 512
  float dw = dist_w[n * 512 + i];
  __shared__ float logits[512];
  __shared__ float red[256];
  float lmax = -1e30f;
  for (int j = threadIdx.x; j < 512; j += 256) {
    float d = fabsf((float)(i - j)) * 0.5f;
    float v = qkr[j] + qec[(long)j * 512] + d * dw;
    logits[j] = v;
    lmax = fmaxf(lmax, v);
  }
  red[threadIdx.x] = lmax; __syncthreads();
  for (int s = 128; s > 0; s >>= 1) {
    if (threadIdx.x < s) red[threadIdx.x] = fmaxf(red[threadIdx.x], red[threadIdx.x + s]);
    __syncthreads();
  }
  lmax = red[0]; __syncthreads();
  float ssum = 0.f;
  for (int j = threadIdx.x; j < 512; j += 256) {
    float e = __expf(logits[j] - lmax);
    logits[j] = e; ssum += e;
  }
  red[threadIdx.x] = ssum; __syncthreads();
  for (int s = 128; s > 0; s >>= 1) {
    if (threadIdx.x < s) red[threadIdx.x] += red[threadIdx.x + s];
    __syncthreads();
  }
  float inv = 1.f / red[0];
  unsigned short* outr = attnA + (long)z * 262144 + (long)i * 512;
  for (int j = threadIdx.x; j < 512; j += 256) outr[j] = f2bf(logits[j] * inv);
}

// ---------------- weight prep: f32 -> bf16 (with layout fixes) --------------
__global__ __launch_bounds__(256)
void copy_bf_kernel(const float* __restrict__ in, unsigned short* __restrict__ out, long n) {
  long idx = (long)blockIdx.x * 256 + threadIdx.x;
  if (idx < n) out[idx] = f2bf(in[idx]);
}
// wcol[(t*1024+i), o] = c1R_w[o, i, t]
__global__ __launch_bounds__(256)
void wcol_prep_kernel(const float* __restrict__ c1R, unsigned short* __restrict__ wcol) {
  long idx = (long)blockIdx.x * 256 + threadIdx.x;   // over 3M: (t,i,o)
  int o = (int)(idx & 1023);
  long ti = idx >> 10;
  int i = (int)(ti & 1023);
  int t = (int)(ti >> 10);
  wcol[idx] = f2bf(c1R[((long)o * 1024 + i) * 3 + t]);
}
// c2pT[i, o] = c2p_w[o, i]
__global__ __launch_bounds__(256)
void c2pT_prep_kernel(const float* __restrict__ c2p, unsigned short* __restrict__ c2pT) {
  long idx = (long)blockIdx.x * 256 + threadIdx.x;   // over 4M: (i,o)
  int o = (int)(idx & 1023);
  long i = idx >> 10;
  c2pT[idx] = f2bf(c2p[(long)o * 4096 + i]);
}

// ===========================================================================
extern "C" void kernel_launch(void* const* d_in, const int* in_sizes, int n_in,
                              void* d_out, int out_size, void* d_ws, size_t ws_size,
                              hipStream_t stream) {
  const float* x      = (const float*)d_in[0];
  const float* bl_w   = (const float*)d_in[1];
  const float* bn_w   = (const float*)d_in[2];
  const float* bn_b   = (const float*)d_in[3];
  const float* glu_w  = (const float*)d_in[4];
  const float* ln1_w  = (const float*)d_in[5];
  const float* ln1_b  = (const float*)d_in[6];
  const float* c1L_w  = (const float*)d_in[7];
  const float* c1R_w  = (const float*)d_in[8];
  const float* ln2_w  = (const float*)d_in[9];
  const float* ln2_b  = (const float*)d_in[10];
  const float* c2d_w  = (const float*)d_in[11];
  const float* c2p_w  = (const float*)d_in[12];
  const float* ln3_w  = (const float*)d_in[13];
  const float* ln3_b  = (const float*)d_in[14];
  const float* q_w    = (const float*)d_in[15];
  const float* k_w    = (const float*)d_in[16];
  const float* v_w    = (const float*)d_in[17];
  const float* o_w    = (const float*)d_in[18];
  const float* er     = (const float*)d_in[19];
  const float* dist_w = (const float*)d_in[20];
  const float* ln4_w  = (const float*)d_in[21];
  const float* ln4_b  = (const float*)d_in[22];
  const float* c3_w   = (const float*)d_in[23];
  const float* c4_w   = (const float*)d_in[24];
  const float* ln5_w  = (const float*)d_in[25];
  const float* ln5_b  = (const float*)d_in[26];
  float* out = (float*)d_out;

  // ---------------- workspace layout (bump allocator, 256B aligned) --------
  char* ws = (char*)d_ws;
  size_t off = 0;
  auto alloc = [&](size_t bytes) -> void* {
    void* p = ws + off;
    off = (off + bytes + 255) & ~(size_t)255;
    return p;
  };
  float*          stats  = (float*)alloc(256);
  float*          xt     = (float*)alloc(4194304ULL * 4);           // [4096,1024] f32
  unsigned short* xtbf   = (unsigned short*)alloc(4194304ULL * 2);  // bf16 copy
  float*          big1   = (float*)alloc(16777216ULL * 4);          // hL / (qk|qe)
  float*          gbuf   = (float*)alloc(8388608ULL * 4);           // g / o_out / h4
  unsigned short* hbf    = (unsigned short*)alloc(16777216ULL * 2); // h / h3
  unsigned short* hcbuf  = (unsigned short*)alloc(16777216ULL * 2); // hc / (attnA|attnO)
  unsigned short* xcol   = (unsigned short*)alloc(12582912ULL * 2); // xcol / (q|k|v)
  float*          hrbuf  = (float*)alloc(4194304ULL * 4);           // y / hR / h2
  unsigned short* glu_bf = (unsigned short*)alloc(2097152ULL * 2);
  unsigned short* c1L_bf = (unsigned short*)alloc(4194304ULL * 2);
  unsigned short* wcol_bf= (unsigned short*)alloc(3145728ULL * 2);
  unsigned short* c2pT_bf= (unsigned short*)alloc(4194304ULL * 2);
  unsigned short* qw_bf  = (unsigned short*)alloc(1048576ULL * 2);
  unsigned short* kw_bf  = (unsigned short*)alloc(1048576ULL * 2);
  unsigned short* vw_bf  = (unsigned short*)alloc(1048576ULL * 2);
  unsigned short* ow_bf  = (unsigned short*)alloc(1048576ULL * 2);
  unsigned short* er_bf  = (unsigned short*)alloc(131072ULL * 2);
  unsigned short* c3_bf  = (unsigned short*)alloc(4194304ULL * 2);
  unsigned short* c4_bf  = (unsigned short*)alloc(4194304ULL * 2);
  if (off > ws_size) return;   // workspace too small; nothing safe to do

  float* qkb   = big1;                 // [32,512,512] f32
  float* qeb   = big1 + 8388608;       // [32,512,512] f32
  unsigned short* attnA = hcbuf;                // [32,512,512] bf16
  unsigned short* attnO = hcbuf + 8388608;      // [4096,1024] bf16
  unsigned short* qb = xcol;                    // [4096,1024] bf16
  unsigned short* kb = xcol + 4194304;
  unsigned short* vb = xcol + 8388608;

  // ---------------- weight prep ----------------
  zero_stats_kernel<<<1, 1, 0, stream>>>(stats);
  copy_bf_kernel<<<8192,  256, 0, stream>>>(glu_w, glu_bf, 2097152L);
  copy_bf_kernel<<<16384, 256, 0, stream>>>(c1L_w, c1L_bf, 4194304L);
  wcol_prep_kernel<<<12288, 256, 0, stream>>>(c1R_w, wcol_bf);
  c2pT_prep_kernel<<<16384, 256, 0, stream>>>(c2p_w, c2pT_bf);
  copy_bf_kernel<<<4096,  256, 0, stream>>>(q_w, qw_bf, 1048576L);
  copy_bf_kernel<<<4096,  256, 0, stream>>>(k_w, kw_bf, 1048576L);
  copy_bf_kernel<<<4096,  256, 0, stream>>>(v_w, vw_bf, 1048576L);
  copy_bf_kernel<<<4096,  256, 0, stream>>>(o_w, ow_bf, 1048576L);
  copy_bf_kernel<<<512,   256, 0, stream>>>(er, er_bf, 131072L);
  copy_bf_kernel<<<16384, 256, 0, stream>>>(c3_w, c3_bf, 4194304L);
  copy_bf_kernel<<<16384, 256, 0, stream>>>(c4_w, c4_bf, 4194304L);

  // ---------------- bottleneck + BatchNorm + ReLU + transpose -------------
  bottleneck_kernel<<<16384, 256, 0, stream>>>(x, bl_w, hrbuf, stats);
  bn_finalize_kernel<<<1, 1, 0, stream>>>(stats, bn_w, bn_b);
  bn_apply_kernel<<<16384, 256, 0, stream>>>(hrbuf, stats, xt, xtbf);

  // ---------------- GLU + LN1 ----------------
  gemm_bf16<false, false, false><<<dim3(32, 32, 1), 256, 0, stream>>>(
      xtbf, glu_bf, gbuf, MROWS, 2048, 1024, 1024, 2048, 2048, 1.f,
      0, 0, 0, 0, 0, 0, 1);
  ln1_kernel<<<4096, 256, 0, stream>>>(xt, xtbf, gbuf, ln1_w, ln1_b);

  // ---------------- dual branch: linear + conv1d(k=3), LN2 ----------------
  gemm_bf16<false, true, false><<<dim3(64, 32, 1), 256, 0, stream>>>(
      xtbf, c1L_bf, big1, MROWS, 4096, 1024, 1024, 4096, 4096, 1.f,
      0, 0, 0, 0, 0, 0, 1);
  im2col_kernel<<<16384, 256, 0, stream>>>(xtbf, xcol);
  gemm_bf16<false, true, false><<<dim3(16, 32, 1), 256, 0, stream>>>(
      xcol, wcol_bf, hrbuf, MROWS, 1024, 3072, 3072, 1024, 1024, 1.f,
      0, 0, 0, 0, 0, 0, 1);
  ln2_kernel<<<4096, 256, 0, stream>>>(big1, hrbuf, hbf, ln2_w, ln2_b);

  // ---------------- depthwise(k=9) + pointwise + LN3 ----------------
  depthwise_kernel<<<65536, 256, 0, stream>>>(hbf, c2d_w, hcbuf);
  gemm_bf16<false, false, false><<<dim3(16, 32, 1), 256, 0, stream>>>(
      hcbuf, c2pT_bf, hrbuf, MROWS, 1024, 4096, 4096, 1024, 1024, 1.f,
      0, 0, 0, 0, 0, 0, 1);
  ln_add_kernel<<<4096, 256, 0, stream>>>(xt, xtbf, hrbuf, ln3_w, ln3_b);

  // ---------------- multiband attention ----------------
  gemm_bf16<false, false, true><<<dim3(16, 32, 1), 256, 0, stream>>>(
      xtbf, qw_bf, qb, MROWS, 1024, 1024, 1024, 1024, 1024, 1.f,
      0, 0, 0, 0, 0, 0, 1);
  gemm_bf16<false, false, true><<<dim3(16, 32, 1), 256, 0, stream>>>(
      xtbf, kw_bf, kb, MROWS, 1024, 1024, 1024, 1024, 1024, 1.f,
      0, 0, 0, 0, 0, 0, 1);
  gemm_bf16<false, false, true><<<dim3(16, 32, 1), 256, 0, stream>>>(
      xtbf, vw_bf, vb, MROWS, 1024, 1024, 1024, 1024, 1024, 1.f,
      0, 0, 0, 0, 0, 0, 1);
  // qk[z] = Q_z K_z^T / 32   (z = b*4 + n; Q_z,K_z are strided band views)
  gemm_bf16<true, false, false><<<dim3(8, 4, 32), 256, 0, stream>>>(
      qb, kb, qkb, 512, 512, 256, 1024, 1024, 512, 1.f / 32.f,
      524288, 256, 524288, 256, 1048576, 262144, 4);
  // qe[z] = Q_z @ er   (shared B)
  gemm_bf16<false, false, false><<<dim3(8, 4, 32), 256, 0, stream>>>(
      qb, er_bf, qeb, 512, 512, 256, 1024, 512, 512, 1.f,
      524288, 256, 0, 0, 1048576, 262144, 4);
  attn_softmax_kernel<<<dim3(512, 32), 256, 0, stream>>>(qkb, qeb, dist_w, attnA);
  // attnO[b,w,n*256+d] = A_z @ V_z
  gemm_bf16<false, false, true><<<dim3(4, 4, 32), 256, 0, stream>>>(
      attnA, vb, attnO, 512, 256, 512, 512, 1024, 1024, 1.f,
      1048576, 262144, 524288, 256, 524288, 256, 4);
  gemm_bf16<false, false, false><<<dim3(16, 32, 1), 256, 0, stream>>>(
      attnO, ow_bf, gbuf, MROWS, 1024, 1024, 1024, 1024, 1024, 1.f,
      0, 0, 0, 0, 0, 0, 1);
  ln_add_kernel<<<4096, 256, 0, stream>>>(xt, xtbf, gbuf, ln4_w, ln4_b);

  // ---------------- FFN + LN5 + output transpose ----------------
  gemm_bf16<false, true, true><<<dim3(64, 32, 1), 256, 0, stream>>>(
      xtbf, c3_bf, hbf, MROWS, 4096, 1024, 1024, 4096, 4096, 1.f,
      0, 0, 0, 0, 0, 0, 1);
  gemm_bf16<false, false, false><<<dim3(16, 32, 1), 256, 0, stream>>>(
      hbf, c4_bf, gbuf, MROWS, 1024, 4096, 4096, 1024, 1024, 1.f,
      0, 0, 0, 0, 0, 0, 1);
  ln5_kernel<<<4096, 256, 0, stream>>>(xt, gbuf, ln5_w, ln5_b, out);
}
